// AttentionFrame_7808250544672
// MI455X (gfx1250) — matrix operand order repaired and checked by reading, not compiled, not run on hardware
//
#include <hip/hip_runtime.h>

// ---- problem constants -----------------------------------------------------
#define HH 48
#define WW 48
#define NHEAD 16
#define DHEAD 64
#define DD (NHEAD * DHEAD)   // 1024
#define HWQ (HH * WW)        // 2304

typedef _Float16 f16;
typedef __attribute__((ext_vector_type(16))) _Float16 v16h;
typedef __attribute__((ext_vector_type(8)))  _Float16 v8h;
typedef __attribute__((ext_vector_type(8)))  float    v8f;
typedef __attribute__((ext_vector_type(4)))  float    v4f;
typedef __attribute__((ext_vector_type(8)))  int      v8i;

// ---- kernel 1: fp32 -> f16 staging (q scaled by DH^-0.5 * log2e; V transposed)
__global__ __launch_bounds__(256) void prep_qkv_kernel(
    const float* __restrict__ q, const float* __restrict__ k, const float* __restrict__ v,
    f16* __restrict__ qf, f16* __restrict__ kf, f16* __restrict__ vT) {
  int tid = blockIdx.x * 256 + threadIdx.x;
  if (tid >= HWQ * DD) return;
  int qi = tid / DD;
  int chn = tid - qi * DD;
  int n = chn >> 6;
  int c = chn & 63;
  const float QSCALE = 0.125f * 1.44269504088896f;  // DH^-0.5 * log2(e)
  qf[((size_t)n * HWQ + qi) * DHEAD + c] = (f16)(q[tid] * QSCALE);
  kf[((size_t)n * HWQ + qi) * DHEAD + c] = (f16)k[tid];
  vT[((size_t)n * DHEAD + c) * HWQ + qi] = (f16)v[tid];
}

// ---- kernel 2: decomposed rel-pos dot products (scaled by log2e) -----------
__global__ __launch_bounds__(256) void prep_rel_kernel(
    const float* __restrict__ q, const float* __restrict__ rph, const float* __restrict__ rpw,
    float* __restrict__ relh, float* __restrict__ relw) {
  int tid = blockIdx.x * 256 + threadIdx.x;
  if (tid >= NHEAD * HWQ * 48) return;
  int j  = tid % 48;
  int t  = tid / 48;
  int qi = t % HWQ;
  int n  = t / HWQ;
  int h = qi / WW;
  int w = qi - h * WW;
  const v4f* qv = (const v4f*)(q + (size_t)qi * DD + n * DHEAD);
  const v4f* rh = (const v4f*)(rph + (size_t)(h - j + 47) * DHEAD);
  const v4f* rw = (const v4f*)(rpw + (size_t)(w - j + 47) * DHEAD);
  v4f ah = {0.f, 0.f, 0.f, 0.f}, aw = {0.f, 0.f, 0.f, 0.f};
#pragma unroll
  for (int i = 0; i < DHEAD / 4; ++i) {
    v4f qq = qv[i];
    ah += qq * rh[i];
    aw += qq * rw[i];
  }
  const float L2E = 1.44269504088896f;
  relh[tid] = (ah[0] + ah[1] + ah[2] + ah[3]) * L2E;
  relw[tid] = (aw[0] + aw[1] + aw[2] + aw[3]) * L2E;
}

__device__ __forceinline__ v8f wmma_f16(v16h a, v16h b, v8f c) {
  return __builtin_amdgcn_wmma_f32_16x16x32_f16(false, a, false, b, (short)0, c, false, false);
}

// assemble a 16-half A-fragment from the two per-lane 16B chunks
__device__ __forceinline__ v16h mk_afrag(const f16* row, int hi) {
  v8h a0 = *(const v8h*)(row + 16 * hi);
  v8h a1 = *(const v8h*)(row + 16 * hi + 8);
  v16h r;
#pragma unroll
  for (int i = 0; i < 8; ++i) { r[i] = a0[i]; r[i + 8] = a1[i]; }
  return r;
}

// ---- kernel 3: transposed flash attention, 1 wave = 16 queries -------------
// S^T = K Q^T  (A = K tile, B = Q^T);   O^T = V^T P^T  (A = V^T, B = P^T)
// => one softmax row (query) = one lane; reductions are in-lane + one xor16.
__global__ __launch_bounds__(128) void attn_kernel(
    const f16* __restrict__ qf, const f16* __restrict__ kf, const f16* __restrict__ vT,
    const float* __restrict__ relh_g, const float* __restrict__ relw_g,
    float* __restrict__ out) {
  __shared__ float s_relh[4][16 * 48];   // 12 KB: rel_h rows for this q-tile

  const int wave = threadIdx.x >> 5;
  const int lane = threadIdx.x & 31;
  const int ln   = lane & 15;      // query column (N of S^T / O^T)
  const int hi   = lane >> 4;      // half-wave selector
  const int head = blockIdx.x / 36;
  const int q0   = (blockIdx.x % 36) * 64 + wave * 16;

  {
    const float* src = relh_g + ((size_t)head * HWQ + q0) * 48;
    float* dst = s_relh[wave];
    for (int i = lane; i < 16 * 48; i += 32) dst[i] = src[i];
  }
  __syncthreads();

  // Q^T B-fragments: lane=query ln, K = d = 32*d2 + 16*hi + i (contiguous 32B)
  v16h qb[2];
  {
    const f16* qrow = qf + ((size_t)head * HWQ + q0 + ln) * DHEAD + 16 * hi;
    qb[0] = *(const v16h*)(qrow);
    qb[1] = *(const v16h*)(qrow + 32);
  }

  // rel_w fragments in S^T C-layout: rwf[t][j] = relw[q=ln][16t + 8hi + j]
  v8f rwf0, rwf1, rwf2;
  {
    const float* rwrow = relw_g + ((size_t)head * HWQ + q0 + ln) * 48 + 8 * hi;
    rwf0 = *(const v8f*)(rwrow);
    rwf1 = *(const v8f*)(rwrow + 16);
    rwf2 = *(const v8f*)(rwrow + 32);
  }

  v8f o[4];
#pragma unroll
  for (int f = 0; f < 4; ++f)
#pragma unroll
    for (int j = 0; j < 8; ++j) o[f][j] = 0.0f;
  float m = -1e30f, l = 0.0f;

  const float* rh = s_relh[wave] + ln * 48;   // per-query rel_h row

  // 96 keys per outer iteration (3 x 32-key chunks) => kw-tile index static
  for (int cc = 0; cc < HWQ / 96; ++cc) {
#pragma unroll
    for (int u = 0; u < 3; ++u) {
      const int chk = cc * 3 + u;
      const int k0  = chk * 32;

      // prefetch next chunk's K/V stream into near caches
      {
        const int kn = (chk + 1 < HWQ / 32) ? k0 + 32 : k0;
        __builtin_prefetch(kf + ((size_t)head * HWQ + kn + ln) * DHEAD, 0, 0);
        __builtin_prefetch(vT + ((size_t)head * DHEAD + ln) * HWQ + kn, 0, 0);
      }

      // ---- S^T tiles: two 16(key)x16(query) tiles ----
      v16h ka[2][2];   // [key-tile][d-chunk]
#pragma unroll
      for (int t2 = 0; t2 < 2; ++t2) {
        const f16* krow = kf + ((size_t)head * HWQ + k0 + 16 * t2 + ln) * DHEAD;
        ka[t2][0] = mk_afrag(krow, hi);
        ka[t2][1] = mk_afrag(krow + 32, hi);
      }
      v8f st0 = {0.f,0.f,0.f,0.f,0.f,0.f,0.f,0.f};
      v8f st1 = {0.f,0.f,0.f,0.f,0.f,0.f,0.f,0.f};
      st0 = wmma_f16(ka[0][0], qb[0], st0);
      st1 = wmma_f16(ka[1][0], qb[0], st1);
      st0 = wmma_f16(ka[0][1], qb[1], st0);
      st1 = wmma_f16(ka[1][1], qb[1], st1);

      // ---- bias: rel_h is scalar per (query, key-row); rel_w vector ----
      const int kh0 = 2 * cc + ((2 * u + 0) >= 3 ? 1 : 0);
      const int kh1 = 2 * cc + ((2 * u + 1) >= 3 ? 1 : 0);
      const int tw0 = (2 * u + 0) % 3;
      const int tw1 = (2 * u + 1) % 3;
      const v8f rwt0 = (tw0 == 0) ? rwf0 : (tw0 == 1) ? rwf1 : rwf2;
      const v8f rwt1 = (tw1 == 0) ? rwf0 : (tw1 == 1) ? rwf1 : rwf2;
      st0 = st0 + rwt0 + rh[kh0];
      st1 = st1 + rwt1 + rh[kh1];

      // ---- online softmax (per-lane scalar state) ----
      v8f tm;
#pragma unroll
      for (int j = 0; j < 8; ++j) tm[j] = fmaxf(st0[j], st1[j]);
      float mv = fmaxf(fmaxf(fmaxf(tm[0], tm[1]), fmaxf(tm[2], tm[3])),
                       fmaxf(fmaxf(tm[4], tm[5]), fmaxf(tm[6], tm[7])));
      mv = fmaxf(mv, __shfl_xor(mv, 16, 32));
      const float nm = fmaxf(m, mv);
      const float cf = __builtin_amdgcn_exp2f(m - nm);
      m = nm;
#pragma unroll
      for (int j = 0; j < 8; ++j) {
        st0[j] = __builtin_amdgcn_exp2f(st0[j] - nm);
        st1[j] = __builtin_amdgcn_exp2f(st1[j] - nm);
      }
      v8f ts = st0 + st1;
      float r = (ts[0] + ts[1]) + (ts[2] + ts[3]) + (ts[4] + ts[5]) + (ts[6] + ts[7]);
      r += __shfl_xor(r, 16, 32);
      l = l * cf + r;

      // ---- P^T C-layout -> B-fragment: pack f16x2 + lane^16 exchange ----
      int pk0[4], pk1[4];
#pragma unroll
      for (int rr = 0; rr < 4; ++rr) {
        pk0[rr] = __builtin_bit_cast(int, __builtin_amdgcn_cvt_pkrtz(st0[2*rr], st0[2*rr+1]));
        pk1[rr] = __builtin_bit_cast(int, __builtin_amdgcn_cvt_pkrtz(st1[2*rr], st1[2*rr+1]));
      }
      v8i pbi;
#pragma unroll
      for (int rr = 0; rr < 4; ++rr) {
        const int send = hi ? pk0[rr] : pk1[rr];
        const int recv = __shfl_xor(send, 16, 32);
        pbi[rr]     = hi ? recv    : pk0[rr];   // chunk keys [16*hi .. 16*hi+7]
        pbi[rr + 4] = hi ? pk1[rr] : recv;      // chunk keys [16*hi+8 .. +15]
      }
      const v16h pb = __builtin_bit_cast(v16h, pbi);

      // ---- O^T = O^T * cf + V^T P^T (4 d-tiles) ----
#pragma unroll
      for (int f = 0; f < 4; ++f) {
        const f16* vrow = vT + ((size_t)head * DHEAD + 16 * f + ln) * HWQ + k0;
        const v16h va = mk_afrag(vrow, hi);
        o[f] = wmma_f16(va, pb, o[f] * cf);
      }
    }
  }

  // ---- epilogue: per-lane contiguous v8f stores (out[q][head*64 + d]) ----
  const float inv = 1.0f / l;
  float* orow = out + (size_t)(q0 + ln) * DD + head * DHEAD + 8 * hi;
#pragma unroll
  for (int f = 0; f < 4; ++f)
    *(v8f*)(orow + 16 * f) = o[f] * inv;
}

// ---- launcher --------------------------------------------------------------
extern "C" void kernel_launch(void* const* d_in, const int* in_sizes, int n_in,
                              void* d_out, int out_size, void* d_ws, size_t ws_size,
                              hipStream_t stream) {
  const float* q   = (const float*)d_in[0];
  const float* k   = (const float*)d_in[1];
  const float* v   = (const float*)d_in[2];
  const float* rph = (const float*)d_in[3];
  const float* rpw = (const float*)d_in[4];
  float* out = (float*)d_out;

  // workspace layout (~28 MB total)
  char* ws = (char*)d_ws;
  size_t off = 0;
  f16* qf = (f16*)(ws + off); off += (size_t)NHEAD * HWQ * DHEAD * sizeof(f16);
  f16* kf = (f16*)(ws + off); off += (size_t)NHEAD * HWQ * DHEAD * sizeof(f16);
  f16* vT = (f16*)(ws + off); off += (size_t)NHEAD * HWQ * DHEAD * sizeof(f16);
  float* relh = (float*)(ws + off); off += (size_t)NHEAD * HWQ * 48 * sizeof(float);
  float* relw = (float*)(ws + off); off += (size_t)NHEAD * HWQ * 48 * sizeof(float);
  (void)off; (void)ws_size; (void)in_sizes; (void)n_in; (void)out_size;

  prep_qkv_kernel<<<(HWQ * DD) / 256, 256, 0, stream>>>(q, k, v, qf, kf, vT);
  prep_rel_kernel<<<(NHEAD * HWQ * 48) / 256, 256, 0, stream>>>(q, rph, rpw, relh, relw);
  attn_kernel<<<NHEAD * (HWQ / 64), 128, 0, stream>>>(qf, kf, vT, relh, relw, out);
}
